// Head_24850680774786
// MI455X (gfx1250) — matrix operand order repaired and checked by reading, not compiled, not run on hardware
//
#include <hip/hip_runtime.h>

#define B_  1024
#define T_  256
#define E_  384
#define H_  64

typedef _Float16 v16h __attribute__((ext_vector_type(16)));
typedef _Float16 v8h  __attribute__((ext_vector_type(8)));
typedef _Float16 v4h  __attribute__((ext_vector_type(4)));
typedef float    v8f  __attribute__((ext_vector_type(8)));
typedef float    v4f  __attribute__((ext_vector_type(4)));

// LDS layout (bytes):
//   Qs [T][H] f16   : 0      .. 32768
//   Ks [T][H] f16   : 32768  .. 65536
//   Vt [H][T] f16   : 65536  .. 98304
//   Pb 8 x [16][32] : 98304  .. 106496
//   Wl [H][E] f16   : 106496 .. 155648   (one weight, staged per phase-1 pass)
#define OFF_KS 32768
#define OFF_VT 65536
#define OFF_PB 98304
#define OFF_WL 106496
#define SMEM_BYTES (OFF_WL + H_*E_*2)

static __device__ __forceinline__ float rmax16(float v) {
  v = fmaxf(v, __shfl_xor(v, 1, 32));
  v = fmaxf(v, __shfl_xor(v, 2, 32));
  v = fmaxf(v, __shfl_xor(v, 4, 32));
  v = fmaxf(v, __shfl_xor(v, 8, 32));
  return v;
}
static __device__ __forceinline__ float rsum16(float v) {
  v += __shfl_xor(v, 1, 32);
  v += __shfl_xor(v, 2, 32);
  v += __shfl_xor(v, 4, 32);
  v += __shfl_xor(v, 8, 32);
  return v;
}
static __device__ __forceinline__ v16h cat8(v8h lo, v8h hi) {
  return __builtin_shufflevector(lo, hi, 0,1,2,3,4,5,6,7,8,9,10,11,12,13,14,15);
}
// convert 16 fp32 (two 16B-aligned chunks of 8, `gap` floats apart) to v16h
static __device__ __forceinline__ v16h cvt16(const float* p, int gap) {
  v4f x0 = *(const v4f*)(p);
  v4f x1 = *(const v4f*)(p + 4);
  v4f x2 = *(const v4f*)(p + gap);
  v4f x3 = *(const v4f*)(p + gap + 4);
  v16h a;
  #pragma unroll
  for (int j = 0; j < 4; ++j) {
    a[j]      = (_Float16)x0[j];
    a[4 + j]  = (_Float16)x1[j];
    a[8 + j]  = (_Float16)x2[j];
    a[12 + j] = (_Float16)x3[j];
  }
  return a;
}
// 16 contiguous f16 from LDS
static __device__ __forceinline__ v16h ld16(const _Float16* p) {
  return cat8(*(const v8h*)p, *(const v8h*)(p + 8));
}

__global__ void __launch_bounds__(256)
fused_head_kernel(const float* __restrict__ embed,
                  const float* __restrict__ Wk,
                  const float* __restrict__ Wq,
                  const float* __restrict__ Wv,
                  float* __restrict__ out) {
  extern __shared__ char smem[];
  _Float16* Qs = (_Float16*)(smem);
  _Float16* Ks = (_Float16*)(smem + OFF_KS);
  _Float16* Vt = (_Float16*)(smem + OFF_VT);
  _Float16* Pb = (_Float16*)(smem + OFF_PB);
  _Float16* Wl = (_Float16*)(smem + OFF_WL);

  const int b    = blockIdx.x;
  const int tid  = threadIdx.x;
  const int wave = tid >> 5;
  const int lane = tid & 31;
  const int ln   = lane & 15;   // position within 16-lane half
  const int lh   = lane >> 4;   // which half

  const v8f zf = {0.f,0.f,0.f,0.f,0.f,0.f,0.f,0.f};

  // ---------------- Phase 1: K/Q/V = embed @ W^T (f16 WMMA, f32 accum) ------
  const float* arowB = embed + ((size_t)b*T_ + (size_t)ln)*E_ + lh*16;
  #pragma unroll 1
  for (int w3 = 0; w3 < 3; ++w3) {
    const float* W = (w3 == 0) ? Wk : ((w3 == 1) ? Wq : Wv);
    __syncthreads();   // everyone done reading Wl from previous pass
    // cooperative fp32 -> f16 conversion of one 64x384 weight into LDS
    #pragma unroll 1
    for (int i = 0; i < 24; ++i) {
      const int e4 = (i*256 + tid) * 4;   // float4 granule
      v4f x = *(const v4f*)(W + e4);
      v4h h;
      #pragma unroll
      for (int j = 0; j < 4; ++j) h[j] = (_Float16)x[j];
      *(v4h*)(Wl + e4) = h;
    }
    __syncthreads();

    #pragma unroll 1
    for (int mt = wave; mt < 16; mt += 8) {
      const float* arow = arowB + (size_t)(mt*16)*E_;
      const _Float16* wrow = Wl + ln*E_ + lh*16;
      v8f c0 = zf, c1 = zf, c2 = zf, c3 = zf;
      #pragma unroll 1
      for (int ks = 0; ks < 12; ++ks) {
        // A 16x32 f16: lane row = ln, 16 contiguous k at lh*16
        v16h aF = cvt16(arow + ks*32, 8);
        // B 32x16 f16: lane col = nt*16+ln, 16 contiguous k at lh*16
        v16h bF0 = ld16(wrow + ks*32);
        c0 = __builtin_amdgcn_wmma_f32_16x16x32_f16(false, aF, false, bF0, (short)0, c0, false, false);
        v16h bF1 = ld16(wrow + 16*E_ + ks*32);
        c1 = __builtin_amdgcn_wmma_f32_16x16x32_f16(false, aF, false, bF1, (short)0, c1, false, false);
        v16h bF2 = ld16(wrow + 32*E_ + ks*32);
        c2 = __builtin_amdgcn_wmma_f32_16x16x32_f16(false, aF, false, bF2, (short)0, c2, false, false);
        v16h bF3 = ld16(wrow + 48*E_ + ks*32);
        c3 = __builtin_amdgcn_wmma_f32_16x16x32_f16(false, aF, false, bF3, (short)0, c3, false, false);
      }
      // C layout: lane l, vgpr r -> row m = r + 8*(l/16), col n = l%16
      const int tbase = mt*16 + 8*lh;
      if (w3 == 0) {
        #pragma unroll
        for (int r = 0; r < 8; ++r) {
          int t = tbase + r;
          Ks[t*H_ +      ln] = (_Float16)c0[r];
          Ks[t*H_ + 16 + ln] = (_Float16)c1[r];
          Ks[t*H_ + 32 + ln] = (_Float16)c2[r];
          Ks[t*H_ + 48 + ln] = (_Float16)c3[r];
        }
      } else if (w3 == 1) {
        #pragma unroll
        for (int r = 0; r < 8; ++r) {
          int t = tbase + r;
          Qs[t*H_ +      ln] = (_Float16)c0[r];
          Qs[t*H_ + 16 + ln] = (_Float16)c1[r];
          Qs[t*H_ + 32 + ln] = (_Float16)c2[r];
          Qs[t*H_ + 48 + ln] = (_Float16)c3[r];
        }
      } else {
        #pragma unroll
        for (int r = 0; r < 8; ++r) {
          int t = tbase + r;
          Vt[(     ln)*T_ + t] = (_Float16)c0[r];   // V stored transposed [H][T]
          Vt[(16 + ln)*T_ + t] = (_Float16)c1[r];
          Vt[(32 + ln)*T_ + t] = (_Float16)c2[r];
          Vt[(48 + ln)*T_ + t] = (_Float16)c3[r];
        }
      }
    }
  }
  __syncthreads();

  // ---------------- Phase 2: causal flash attention per wave ----------------
  _Float16* Pw = Pb + wave*512;        // private 16x32 f16 staging
  const float scale = 0.125f;          // 64^-0.5
  const float NEG   = -1e30f;

  #pragma unroll 1
  for (int qb = wave; qb < 16; qb += 8) {
    const int t0 = qb*16;
    // Q A-fragments for h = 0..31 and 32..63
    v16h qa0, qa1;
    {
      const _Float16* qp = Qs + (t0 + ln)*H_ + lh*16;
      qa0 = cat8(*(const v8h*)qp,        *(const v8h*)(qp + 8));
      qa1 = cat8(*(const v8h*)(qp + 32), *(const v8h*)(qp + 40));
    }
    v8f acc[4] = { zf, zf, zf, zf };
    float mrow[8], lrow[8];
    #pragma unroll
    for (int r = 0; r < 8; ++r) { mrow[r] = NEG; lrow[r] = 0.f; }

    const int kbmax = (t0 + 15) >> 5;          // causal: skip fully masked blocks
    #pragma unroll 1
    for (int kb = 0; kb <= kbmax; ++kb) {
      const int s0 = kb*32;
      // scores: S = Q(16x64) @ K^T(64x32): two 16x16 n-tiles x two 32-wide k-chunks
      v8f sl = zf, sh2 = zf;
      {
        const _Float16* kp0 = Ks + (s0 + ln)*H_ + lh*16;
        sl  = __builtin_amdgcn_wmma_f32_16x16x32_f16(false, qa0, false, ld16(kp0),      (short)0, sl,  false, false);
        sl  = __builtin_amdgcn_wmma_f32_16x16x32_f16(false, qa1, false, ld16(kp0 + 32), (short)0, sl,  false, false);
        const _Float16* kp1 = kp0 + 16*H_;
        sh2 = __builtin_amdgcn_wmma_f32_16x16x32_f16(false, qa0, false, ld16(kp1),      (short)0, sh2, false, false);
        sh2 = __builtin_amdgcn_wmma_f32_16x16x32_f16(false, qa1, false, ld16(kp1 + 32), (short)0, sh2, false, false);
      }
      const bool needMask = (s0 + 31 > t0);
      // online softmax update (row stats via 16-lane butterflies)
      #pragma unroll
      for (int r = 0; r < 8; ++r) {
        const int t = t0 + r + 8*lh;
        float a0 = sl[r]  * scale;
        float a1 = sh2[r] * scale;
        if (needMask) {
          if (s0 + ln > t)      a0 = NEG;
          if (s0 + 16 + ln > t) a1 = NEG;
        }
        float mx    = rmax16(fmaxf(a0, a1));
        float mnew  = fmaxf(mrow[r], mx);
        float alpha = __expf(mrow[r] - mnew);
        mrow[r] = mnew;
        float p0 = __expf(a0 - mnew);
        float p1 = __expf(a1 - mnew);
        lrow[r] = lrow[r]*alpha + rsum16(p0 + p1);
        #pragma unroll
        for (int ht = 0; ht < 4; ++ht) acc[ht][r] *= alpha;
        // stage P in A-friendly [m][s] layout
        Pw[(r + 8*lh)*32 + ln]      = (_Float16)p0;
        Pw[(r + 8*lh)*32 + 16 + ln] = (_Float16)p1;
      }
      // O += P(16x32) @ V(32x64)
      const _Float16* pp = Pw + ln*32 + lh*16;
      v16h pa = cat8(*(const v8h*)pp, *(const v8h*)(pp + 8));
      #pragma unroll
      for (int ht = 0; ht < 4; ++ht) {
        const _Float16* vp = Vt + (ht*16 + ln)*T_ + s0 + lh*16;
        acc[ht] = __builtin_amdgcn_wmma_f32_16x16x32_f16(false, pa, false, ld16(vp), (short)0, acc[ht], false, false);
      }
    }
    // normalize + store
    #pragma unroll
    for (int r = 0; r < 8; ++r) {
      const int t = t0 + r + 8*lh;
      const float inv = 1.0f / lrow[r];
      float* op = out + ((size_t)b*T_ + t)*H_ + ln;
      op[0]  = acc[0][r] * inv;
      op[16] = acc[1][r] * inv;
      op[32] = acc[2][r] * inv;
      op[48] = acc[3][r] * inv;
    }
  }
}

extern "C" void kernel_launch(void* const* d_in, const int* in_sizes, int n_in,
                              void* d_out, int out_size, void* d_ws, size_t ws_size,
                              hipStream_t stream) {
  (void)in_sizes; (void)n_in; (void)out_size; (void)d_ws; (void)ws_size;
  const float* embed = (const float*)d_in[0];
  const float* Wk    = (const float*)d_in[1];
  const float* Wq    = (const float*)d_in[2];
  const float* Wv    = (const float*)d_in[3];
  float* out = (float*)d_out;

  (void)hipFuncSetAttribute((const void*)fused_head_kernel,
                            hipFuncAttributeMaxDynamicSharedMemorySize, SMEM_BYTES);
  fused_head_kernel<<<dim3(B_), dim3(256), SMEM_BYTES, stream>>>(embed, Wk, Wq, Wv, out);
}
